// Spatio_Temporal_Set_Attention_25039659335844
// MI455X (gfx1250) — compile-verified, matrix operand order checked by reading, not run on hardware
//
#include <hip/hip_runtime.h>
#include <math.h>

#define DD   768
#define HD   1024
#define NSLOT 8
#define NP   196
#define NSQ  197
#define NOUT 128
#define NBS  32
#define ROWS_T (NBS*NSQ)   /* 6304 */
#define ROWS_P (NBS*NP)    /* 6272 */
#define G3   (3*DD)        /* 2304 */
#define RC   784           /* sisa chunk (patch-seqs) */
#define NCHUNK 8

typedef _Float16 v16h __attribute__((ext_vector_type(16)));
typedef _Float16 v8h  __attribute__((ext_vector_type(8)));
typedef _Float16 v4h  __attribute__((ext_vector_type(4)));
typedef float    v8f  __attribute__((ext_vector_type(8)));

enum { ACT_NONE=0, ACT_RELU=1, ACT_GELU=2, ACT_TANH=3 };

__device__ __forceinline__ float act_apply(float x, int act) {
  switch (act) {
    case ACT_RELU: return x > 0.f ? x : 0.f;
    case ACT_GELU: {
      float x3 = x*x*x;
      return 0.5f*x*(1.f + tanhf(0.7978845608028654f*(x + 0.044715f*x3)));
    }
    case ACT_TANH: return tanhf(x);
    default: return x;
  }
}

__device__ __forceinline__ float wred(float s) {
  #pragma unroll
  for (int m = 16; m; m >>= 1) s += __shfl_xor(s, m, 32);
  return s;
}

__device__ __forceinline__ v4h cvt4(const float4 f) {
  v4h h;
  h[0] = (_Float16)f.x; h[1] = (_Float16)f.y;
  h[2] = (_Float16)f.z; h[3] = (_Float16)f.w;
  return h;
}

__device__ __forceinline__ v16h frag_ld(const _Float16* base, int row, int half) {
  const v8h lo = *reinterpret_cast<const v8h*>(base + row*40 + half*8);
  const v8h hi = *reinterpret_cast<const v8h*>(base + row*40 + 16 + half*8);
  v16h f;
  #pragma unroll
  for (int e = 0; e < 8; ++e) { f[e] = lo[e]; f[8+e] = hi[e]; }
  return f;
}

// ======================= WMMA GEMM =======================
// C = act(A@B + bias) + add[(row/add_div)*N + col]
// A (M,K) row-major lda. B: TRANSB? (N,K) ldb=K : (K,N) ldb=N.
// 128x64 C-tile / 256 threads; 8 waves in 4x2; each wave 32x32 = 4 wmma/K-step.
// Full interior blocks run a software-pipelined loop with double-buffered LDS
// (1 barrier / K-step); edges fall back to a guarded 2-barrier loop.
template<int TRANSB>
__global__ __launch_bounds__(256)
void gemm_wmma(const float* __restrict__ A, int lda,
               const float* __restrict__ B, int ldb,
               const float* __restrict__ bias,
               const float* __restrict__ add, int add_div,
               float* __restrict__ C, int ldc,
               int M, int N, int K, int act)
{
  __shared__ __align__(16) _Float16 As[2][128*40];
  __shared__ __align__(16) _Float16 Bs[2][64*40];   // [n][k]
  const int tid  = threadIdx.x;
  const int lane = tid & 31;
  const int wave = tid >> 5;
  const int wr = (wave >> 1) * 32;    // 0,32,64,96
  const int wc = (wave & 1) * 32;     // 0 or 32
  const int bm = blockIdx.y * 128;
  const int bn = blockIdx.x * 64;
  const int half = lane >> 4;
  const int fr = lane & 15;

  const bool fullA = (bm + 128 <= M) && ((lda & 3) == 0);
  const bool fullB = (bn + 64 <= N) && ((ldb & 3) == 0);
  const bool pipe  = fullA && fullB && ((K & 31) == 0);

  v8f c00 = {}, c01 = {}, c10 = {}, c11 = {};
  const int kt = (K + 31) >> 5;

  if (pipe) {
    float4 ra[4], rb[2];
    // ---- stage tile 0 ----
    #pragma unroll
    for (int s = 0; s < 4; ++s) {
      int i = tid + s*256;
      ra[s] = *reinterpret_cast<const float4*>(&A[(long)(bm + (i>>3))*lda + (i&7)*4]);
    }
    if (TRANSB) {
      #pragma unroll
      for (int s = 0; s < 2; ++s) {
        int i = tid + s*256;
        rb[s] = *reinterpret_cast<const float4*>(&B[(long)(bn + (i>>3))*ldb + (i&7)*4]);
      }
    } else {
      #pragma unroll
      for (int s = 0; s < 2; ++s) {
        int i = tid + s*256;
        rb[s] = *reinterpret_cast<const float4*>(&B[(long)(i>>4)*ldb + bn + (i&15)*4]);
      }
    }
    #pragma unroll
    for (int s = 0; s < 4; ++s) {
      int i = tid + s*256;
      *reinterpret_cast<v4h*>(&As[0][(i>>3)*40 + (i&7)*4]) = cvt4(ra[s]);
    }
    if (TRANSB) {
      #pragma unroll
      for (int s = 0; s < 2; ++s) {
        int i = tid + s*256;
        *reinterpret_cast<v4h*>(&Bs[0][(i>>3)*40 + (i&7)*4]) = cvt4(rb[s]);
      }
    } else {
      #pragma unroll
      for (int s = 0; s < 2; ++s) {
        int i = tid + s*256;
        int kr = i >> 4, n0 = (i & 15)*4;
        Bs[0][(n0+0)*40 + kr] = (_Float16)rb[s].x;
        Bs[0][(n0+1)*40 + kr] = (_Float16)rb[s].y;
        Bs[0][(n0+2)*40 + kr] = (_Float16)rb[s].z;
        Bs[0][(n0+3)*40 + kr] = (_Float16)rb[s].w;
      }
    }
    __syncthreads();

    for (int k0 = 0; k0 < kt; ++k0) {
      const int cur = k0 & 1, nxt = cur ^ 1;
      const bool more = (k0 + 1 < kt);
      const int kb = (k0 + 1) << 5;
      if (more) {   // global loads for tile k0+1 (latency hidden under wmma)
        #pragma unroll
        for (int s = 0; s < 4; ++s) {
          int i = tid + s*256;
          ra[s] = *reinterpret_cast<const float4*>(&A[(long)(bm + (i>>3))*lda + kb + (i&7)*4]);
        }
        if (TRANSB) {
          #pragma unroll
          for (int s = 0; s < 2; ++s) {
            int i = tid + s*256;
            rb[s] = *reinterpret_cast<const float4*>(&B[(long)(bn + (i>>3))*ldb + kb + (i&7)*4]);
          }
        } else {
          #pragma unroll
          for (int s = 0; s < 2; ++s) {
            int i = tid + s*256;
            rb[s] = *reinterpret_cast<const float4*>(&B[(long)(kb + (i>>4))*ldb + bn + (i&15)*4]);
          }
        }
      }
      {
        const v16h a0 = frag_ld(&As[cur][0], wr + fr,      half);
        const v16h a1 = frag_ld(&As[cur][0], wr + 16 + fr, half);
        const v16h b0 = frag_ld(&Bs[cur][0], wc + fr,      half);
        const v16h b1 = frag_ld(&Bs[cur][0], wc + 16 + fr, half);
        c00 = __builtin_amdgcn_wmma_f32_16x16x32_f16(false, a0, false, b0, (short)0, c00, false, false);
        c01 = __builtin_amdgcn_wmma_f32_16x16x32_f16(false, a0, false, b1, (short)0, c01, false, false);
        c10 = __builtin_amdgcn_wmma_f32_16x16x32_f16(false, a1, false, b0, (short)0, c10, false, false);
        c11 = __builtin_amdgcn_wmma_f32_16x16x32_f16(false, a1, false, b1, (short)0, c11, false, false);
      }
      if (more) {
        #pragma unroll
        for (int s = 0; s < 4; ++s) {
          int i = tid + s*256;
          *reinterpret_cast<v4h*>(&As[nxt][(i>>3)*40 + (i&7)*4]) = cvt4(ra[s]);
        }
        if (TRANSB) {
          #pragma unroll
          for (int s = 0; s < 2; ++s) {
            int i = tid + s*256;
            *reinterpret_cast<v4h*>(&Bs[nxt][(i>>3)*40 + (i&7)*4]) = cvt4(rb[s]);
          }
        } else {
          #pragma unroll
          for (int s = 0; s < 2; ++s) {
            int i = tid + s*256;
            int kr = i >> 4, n0 = (i & 15)*4;
            Bs[nxt][(n0+0)*40 + kr] = (_Float16)rb[s].x;
            Bs[nxt][(n0+1)*40 + kr] = (_Float16)rb[s].y;
            Bs[nxt][(n0+2)*40 + kr] = (_Float16)rb[s].z;
            Bs[nxt][(n0+3)*40 + kr] = (_Float16)rb[s].w;
          }
        }
      }
      __syncthreads();
    }
  } else {
    // ---- guarded fallback (edge blocks / ragged K) ----
    for (int k0 = 0; k0 < kt; ++k0) {
      const int kb = k0 << 5;
      #pragma unroll
      for (int i = tid; i < 4096; i += 256) {
        int r = i >> 5, c = i & 31;
        int gr = bm + r, gk = kb + c;
        float v = 0.f;
        if (gr < M && gk < K) v = A[(long)gr*lda + gk];
        As[0][r*40 + c] = (_Float16)v;
      }
      #pragma unroll
      for (int i = tid; i < 2048; i += 256) {
        int n = i >> 5, c = i & 31;
        int gn = bn + n, gk = kb + c;
        float v = 0.f;
        if (gn < N && gk < K) v = TRANSB ? B[(long)gn*ldb + gk] : B[(long)gk*ldb + gn];
        Bs[0][n*40 + c] = (_Float16)v;
      }
      __syncthreads();
      {
        const v16h a0 = frag_ld(&As[0][0], wr + fr,      half);
        const v16h a1 = frag_ld(&As[0][0], wr + 16 + fr, half);
        const v16h b0 = frag_ld(&Bs[0][0], wc + fr,      half);
        const v16h b1 = frag_ld(&Bs[0][0], wc + 16 + fr, half);
        c00 = __builtin_amdgcn_wmma_f32_16x16x32_f16(false, a0, false, b0, (short)0, c00, false, false);
        c01 = __builtin_amdgcn_wmma_f32_16x16x32_f16(false, a0, false, b1, (short)0, c01, false, false);
        c10 = __builtin_amdgcn_wmma_f32_16x16x32_f16(false, a1, false, b0, (short)0, c10, false, false);
        c11 = __builtin_amdgcn_wmma_f32_16x16x32_f16(false, a1, false, b1, (short)0, c11, false, false);
      }
      __syncthreads();
    }
  }

  // ---- epilogue ----
  #pragma unroll
  for (int ti = 0; ti < 2; ++ti) {
    const int r0 = bm + wr + ti*16 + half*8;
    #pragma unroll
    for (int tj = 0; tj < 2; ++tj) {
      const int col = bn + wc + tj*16 + fr;
      if (col < N) {
        const float bv = bias ? bias[col] : 0.f;
        const v8f& acc = ti ? (tj ? c11 : c10) : (tj ? c01 : c00);
        #pragma unroll
        for (int v = 0; v < 8; ++v) {
          int row = r0 + v;
          if (row < M) {
            float val = act_apply(acc[v] + bv, act);
            if (add) val += add[(long)(row/add_div)*N + col];
            C[(long)row*ldc + col] = val;
          }
        }
      }
    }
  }
}

// ======================= elementwise / small kernels =======================
__global__ void patchify_k(const float* __restrict__ x, float* __restrict__ o) {
  long idx = (long)blockIdx.x*256 + threadIdx.x;
  if (idx >= (long)ROWS_P*DD) return;
  int f = idx % DD; long bp = idx / DD;
  int p = bp % NP; int b = bp / NP;
  int c = f % 3; int pix = f / 3; int px = pix & 15; int py = pix >> 4;
  int gx = p % 14; int gy = p / 14;
  int hh = gy*16 + py, ww = gx*16 + px;
  o[idx] = x[(((long)b*3 + c)*224 + hh)*224 + ww];
}

__global__ void assemble_t_k(const float* __restrict__ emb, const float* __restrict__ cls,
                             const float* __restrict__ pos, float* __restrict__ t) {
  long idx = (long)blockIdx.x*256 + threadIdx.x;
  if (idx >= (long)ROWS_T*DD) return;
  int d = idx % DD; long bs = idx / DD;
  int s = bs % NSQ; int b = bs / NSQ;
  float v = (s == 0) ? cls[d] : emb[((long)b*NP + (s-1))*DD + d];
  t[idx] = v + pos[(long)s*DD + d];
}

// one wave per row; d must be a multiple of 128 (768/1024 both are)
__global__ void ln_k(const float* __restrict__ x, const float* __restrict__ g,
                     const float* __restrict__ b, float* __restrict__ y,
                     long rows, int d) {
  long w = ((long)blockIdx.x*blockDim.x + threadIdx.x) >> 5;
  int lane = threadIdx.x & 31;
  if (w >= rows) return;
  const float4* xr = reinterpret_cast<const float4*>(x + w*d);
  const int q = d >> 2;
  float s = 0.f;
  for (int i = lane; i < q; i += 32) {
    float4 f = xr[i]; s += f.x + f.y + f.z + f.w;
  }
  s = wred(s);
  float mean = s / d;
  float v = 0.f;
  for (int i = lane; i < q; i += 32) {
    float4 f = xr[i];
    float a0=f.x-mean, a1=f.y-mean, a2=f.z-mean, a3=f.w-mean;
    v += a0*a0 + a1*a1 + a2*a2 + a3*a3;
  }
  v = wred(v);
  float inv = rsqrtf(v/d + 1e-5f);
  float4* yr = reinterpret_cast<float4*>(y + w*d);
  const float4* gr = reinterpret_cast<const float4*>(g);
  const float4* br = reinterpret_cast<const float4*>(b);
  for (int i = lane; i < q; i += 32) {
    float4 f = xr[i], gg = gr[i], bb = br[i], o;
    o.x = (f.x-mean)*inv*gg.x + bb.x;
    o.y = (f.y-mean)*inv*gg.y + bb.y;
    o.z = (f.z-mean)*inv*gg.z + bb.z;
    o.w = (f.w-mean)*inv*gg.w + bb.w;
    yr[i] = o;
  }
}

__global__ void bcast_row_k(const float* __restrict__ mu, float* __restrict__ o, long n) {
  long idx = (long)blockIdx.x*256 + threadIdx.x;
  if (idx < n) o[idx] = mu[idx % DD];
}

// dots[b][j][n] = 0.125 * <q[b*8+j], k[b*197+n]>   (one wave per output)
__global__ void dots_k(const float* __restrict__ q, const float* __restrict__ kk,
                       float* __restrict__ dots) {
  long w = ((long)blockIdx.x*blockDim.x + threadIdx.x) >> 5;
  int lane = threadIdx.x & 31;
  if (w >= (long)NBS*NSLOT*NSQ) return;
  int n = w % NSQ; long bj = w / NSQ; int b = bj / NSLOT;
  const float4* qr = reinterpret_cast<const float4*>(q + bj*DD);
  const float4* kr = reinterpret_cast<const float4*>(kk + ((long)b*NSQ + n)*DD);
  float s = 0.f;
  for (int i = lane; i < DD/4; i += 32) {
    float4 a = qr[i], c = kr[i];
    s += a.x*c.x + a.y*c.y + a.z*c.z + a.w*c.w;
  }
  s = wred(s);
  if (lane == 0) dots[w] = s * 0.125f;
}

// softmax over the 8 slots per (b,n); +EPS  -> attn_raw
__global__ void softmax_slots_k(const float* __restrict__ dots, float* __restrict__ attn) {
  int t = blockIdx.x*256 + threadIdx.x;
  if (t >= NBS*NSQ) return;
  int n = t % NSQ, b = t / NSQ;
  float vals[NSLOT], m = -1e30f;
  #pragma unroll
  for (int j = 0; j < NSLOT; ++j) {
    vals[j] = dots[((long)b*NSLOT + j)*NSQ + n];
    m = fmaxf(m, vals[j]);
  }
  float sum = 0.f;
  #pragma unroll
  for (int j = 0; j < NSLOT; ++j) { vals[j] = expf(vals[j]-m); sum += vals[j]; }
  #pragma unroll
  for (int j = 0; j < NSLOT; ++j)
    attn[((long)b*NSLOT + j)*NSQ + n] = vals[j]/sum + 1e-8f;
}

// sinv[bj] = 1 / sum_n attn[bj][n]   (one wave per bj)
__global__ void attn_sum_k(const float* __restrict__ attn, float* __restrict__ sinv) {
  int w = (blockIdx.x*blockDim.x + threadIdx.x) >> 5;
  int lane = threadIdx.x & 31;
  if (w >= NBS*NSLOT) return;
  float s = 0.f;
  for (int n = lane; n < NSQ; n += 32) s += attn[(long)w*NSQ + n];
  s = wred(s);
  if (lane == 0) sinv[w] = 1.f / s;
}

// upd[bj][d] = sinv[bj] * sum_n attn[bj][n] * v[b][n][d]
__global__ void upd_k(const float* __restrict__ attn, const float* __restrict__ sinv,
                      const float* __restrict__ v, float* __restrict__ upd) {
  long idx = (long)blockIdx.x*256 + threadIdx.x;
  if (idx >= (long)NBS*NSLOT*DD) return;
  int d = idx % DD; long bj = idx / DD; int b = bj / NSLOT;
  float s = 0.f;
  for (int n = 0; n < NSQ; ++n)
    s += attn[bj*NSQ + n] * v[((long)b*NSQ + n)*DD + d];
  upd[idx] = s * sinv[bj];
}

// GRU gate fusion: h = (1-z)*n + z*h (in place); gi row = r/gi_div
__global__ void gru_k(const float* __restrict__ gi, int gi_div,
                      const float* __restrict__ gh, float* __restrict__ h, long rows) {
  long idx = (long)blockIdx.x*256 + threadIdx.x;
  if (idx >= rows*DD) return;
  int d = idx % DD; long r = idx / DD;
  const float* gir = gi + (r/gi_div)*(long)G3;
  const float* ghr = gh + r*(long)G3;
  float rr = 1.f/(1.f + expf(-(gir[d]        + ghr[d])));
  float zz = 1.f/(1.f + expf(-(gir[DD+d]     + ghr[DD+d])));
  float nn = tanhf(gir[2*DD+d] + rr*ghr[2*DD+d]);
  h[idx] = (1.f-zz)*nn + zz*h[idx];
}

// t[b][n][d] += sum_j attn[bj][n]*sinv[bj]*slots[bj][d]
__global__ void readback_k(const float* __restrict__ attn, const float* __restrict__ sinv,
                           const float* __restrict__ slots, float* __restrict__ t) {
  long idx = (long)blockIdx.x*256 + threadIdx.x;
  if (idx >= (long)ROWS_T*DD) return;
  int d = idx % DD; long bn = idx / DD;
  int n = bn % NSQ; int b = bn / NSQ;
  float s = 0.f;
  #pragma unroll
  for (int j = 0; j < NSLOT; ++j) {
    long bj = (long)b*NSLOT + j;
    s += attn[bj*NSQ + n] * sinv[bj] * slots[bj*DD + d];
  }
  t[idx] += s;
}

// (b,197,768) -> (b,768,197)
__global__ void trans_sd_k(const float* __restrict__ t, float* __restrict__ tt) {
  long idx = (long)blockIdx.x*256 + threadIdx.x;
  if (idx >= (long)NBS*DD*NSQ) return;
  int s = idx % NSQ; long bd = idx / NSQ;
  int d = bd % DD; int b = bd / DD;
  tt[idx] = t[((long)b*NSQ + s)*DD + d];
}

// orix[b][d] = <tmm[b,d,:], msW> + msb   (one wave per (b,d))
__global__ void ms_k(const float* __restrict__ tmm, const float* __restrict__ w,
                     const float* __restrict__ b0, float* __restrict__ orix) {
  long wv = ((long)blockIdx.x*blockDim.x + threadIdx.x) >> 5;
  int lane = threadIdx.x & 31;
  if (wv >= (long)NBS*DD) return;
  float s = 0.f;
  for (int p = lane; p < NP; p += 32) s += tmm[wv*NP + p]*w[p];
  s = wred(s);
  if (lane == 0) orix[wv] = s + b0[0];
}

// (b,768,196) -> (b,196,768)
__global__ void trans_dp_k(const float* __restrict__ tmm, float* __restrict__ xp2) {
  long idx = (long)blockIdx.x*256 + threadIdx.x;
  if (idx >= (long)NBS*NP*DD) return;
  int d = idx % DD; long bp = idx / DD;
  int p = bp % NP; int b = bp / NP;
  xp2[idx] = tmm[((long)b*DD + d)*NP + p];
}

// y[r][d] = x1c[r][d] + sum_j sl[r*8+j][d]
__global__ void slotsum_k(const float* __restrict__ sl, const float* __restrict__ x1c,
                          float* __restrict__ y, long rows) {
  long idx = (long)blockIdx.x*256 + threadIdx.x;
  if (idx >= rows*DD) return;
  int d = idx % DD; long r = idx / DD;
  float s = x1c[idx];
  #pragma unroll
  for (int j = 0; j < NSLOT; ++j) s += sl[(r*NSLOT + j)*(long)DD + d];
  y[idx] = s;
}

// d_out[p][b][o] = ob[(b*196+p)*128+o]
__global__ void outperm_k(const float* __restrict__ ob, float* __restrict__ out) {
  long idx = (long)blockIdx.x*256 + threadIdx.x;
  if (idx >= (long)NP*NBS*NOUT) return;
  int o = idx % NOUT; long pb = idx / NOUT;
  int b = pb % NBS; int p = pb / NBS;
  out[idx] = ob[((long)b*NP + p)*NOUT + o];
}

// ======================= host orchestration =======================
extern "C" void kernel_launch(void* const* d_in, const int* in_sizes, int n_in,
                              void* d_out, int out_size, void* d_ws, size_t ws_size,
                              hipStream_t stream) {
  (void)in_sizes; (void)n_in; (void)out_size; (void)ws_size;
  auto I = [&](int i) { return (const float*)d_in[i]; };

  float* w = (float*)d_ws;
  size_t off = 0;
  auto AL = [&](size_t n) { float* p = w + off; off += n; return p; };

  float* PATCH = AL((size_t)ROWS_P*DD);
  float* EMB   = AL((size_t)ROWS_P*DD);
  float* T     = AL((size_t)ROWS_T*DD);
  float* XLN   = AL((size_t)ROWS_T*DD);
  float* XI    = AL((size_t)ROWS_T*DD);
  float* KB    = AL((size_t)ROWS_T*DD);
  float* VB    = AL((size_t)ROWS_T*DD);
  float* HIDB  = AL((size_t)ROWS_T*HD);
  float* S     = AL((size_t)NBS*NSLOT*DD);
  float* SLN   = AL((size_t)NBS*NSLOT*DD);
  float* Q     = AL((size_t)NBS*NSLOT*DD);
  float* DOTS  = AL((size_t)NBS*NSLOT*NSQ);
  float* ATTN  = AL((size_t)NBS*NSLOT*NSQ);
  float* SINV  = AL((size_t)NBS*NSLOT);
  float* UPD   = AL((size_t)NBS*NSLOT*DD);
  float* GIt   = AL((size_t)NBS*NSLOT*G3);
  float* GHt   = AL((size_t)NBS*NSLOT*G3);
  float* HIDt  = AL((size_t)NBS*NSLOT*HD);
  float* TT    = AL((size_t)NBS*DD*NSQ);
  float* TMM   = AL((size_t)NBS*DD*NP);
  float* ORIX  = AL((size_t)NBS*DD);
  float* XP2   = AL((size_t)ROWS_P*DD);
  float* X1    = AL((size_t)ROWS_P*DD);
  float* XS1   = AL((size_t)ROWS_P*DD);
  float* XS2   = AL((size_t)ROWS_P*DD);
  float* VS    = AL((size_t)ROWS_P*DD);
  float* GIF   = AL((size_t)ROWS_P*G3);
  float* SL    = AL((size_t)RC*NSLOT*DD);
  float* SLNc  = AL((size_t)RC*NSLOT*DD);
  float* GHc   = AL((size_t)RC*NSLOT*G3);
  float* HIDc  = AL((size_t)RC*NSLOT*HD);
  float* Yc    = AL((size_t)RC*DD);
  float* HH    = AL((size_t)RC*HD);
  float* OB    = AL((size_t)ROWS_P*NOUT);

  auto gemm = [&](const float* A, int lda, const float* B, int ldb, int transB,
                  const float* bias, const float* add, int add_div,
                  float* C, int ldc, int M, int N, int K, int act) {
    dim3 g((N+63)/64, (M+127)/128);
    if (transB)
      gemm_wmma<1><<<g, 256, 0, stream>>>(A, lda, B, ldb, bias, add, add_div,
                                          C, ldc, M, N, K, act);
    else
      gemm_wmma<0><<<g, 256, 0, stream>>>(A, lda, B, ldb, bias, add, add_div,
                                          C, ldc, M, N, K, act);
  };
  auto ew = [&](long n) { return dim3((unsigned)((n + 255)/256)); };
  auto ln = [&](const float* x, const float* g, const float* b, float* y, long rows) {
    ln_k<<<dim3((unsigned)((rows+7)/8)), 256, 0, stream>>>(x, g, b, y, rows, DD);
  };

  // ---- patch embed ----
  patchify_k<<<ew((long)ROWS_P*DD), 256, 0, stream>>>(I(0), PATCH);
  gemm(PATCH, DD, I(3), DD, 0, I(4), nullptr, 1, EMB, DD, ROWS_P, DD, DD, ACT_NONE);
  assemble_t_k<<<ew((long)ROWS_T*DD), 256, 0, stream>>>(EMB, I(1), I(2), T);

  // ---- tisa: 4 layers ----
  for (int l = 0; l < 4; ++l) {
    int lb = 5 + 26*l;
    const float *mu=I(lb+2), *Wq=I(lb+7), *Wk=I(lb+8), *Wv=I(lb+9);
    const float *Wih=I(lb+10), *Whh=I(lb+11), *bih=I(lb+12), *bhh=I(lb+13);
    const float *mW1=I(lb+16), *mb1=I(lb+17), *mW2=I(lb+18), *mb2=I(lb+19);

    ln(T, I(lb+0), I(lb+1), XLN, ROWS_T);                 // ln1
    ln(XLN, I(lb+3), I(lb+4), XI, ROWS_T);                // ln_in
    gemm(XI, DD, Wk, DD, 0, nullptr, nullptr, 1, KB, DD, ROWS_T, DD, DD, ACT_NONE);
    gemm(XI, DD, Wv, DD, 0, nullptr, nullptr, 1, VB, DD, ROWS_T, DD, DD, ACT_NONE);
    bcast_row_k<<<ew((long)NBS*NSLOT*DD), 256, 0, stream>>>(mu, S, (long)NBS*NSLOT*DD);

    for (int it = 0; it < 3; ++it) {
      ln(S, I(lb+5), I(lb+6), SLN, NBS*NSLOT);            // ln_s
      gemm(SLN, DD, Wq, DD, 0, nullptr, nullptr, 1, Q, DD, NBS*NSLOT, DD, DD, ACT_NONE);
      dots_k<<<dim3((NBS*NSLOT*NSQ + 7)/8), 256, 0, stream>>>(Q, KB, DOTS);
      softmax_slots_k<<<ew(NBS*NSQ), 256, 0, stream>>>(DOTS, ATTN);
      attn_sum_k<<<dim3((NBS*NSLOT + 7)/8), 256, 0, stream>>>(ATTN, SINV);
      upd_k<<<ew((long)NBS*NSLOT*DD), 256, 0, stream>>>(ATTN, SINV, VB, UPD);
      gemm(UPD, DD, Wih, DD, 1, bih, nullptr, 1, GIt, G3, NBS*NSLOT, G3, DD, ACT_NONE);
      gemm(S,   DD, Whh, DD, 1, bhh, nullptr, 1, GHt, G3, NBS*NSLOT, G3, DD, ACT_NONE);
      gru_k<<<ew((long)NBS*NSLOT*DD), 256, 0, stream>>>(GIt, 1, GHt, S, NBS*NSLOT);
      ln(S, I(lb+14), I(lb+15), SLN, NBS*NSLOT);          // ln_ff
      gemm(SLN, DD, mW1, HD, 0, mb1, nullptr, 1, HIDt, HD, NBS*NSLOT, HD, DD, ACT_RELU);
      gemm(HIDt, HD, mW2, DD, 0, mb2, S, 1, S, DD, NBS*NSLOT, DD, HD, ACT_NONE);
    }
    readback_k<<<ew((long)ROWS_T*DD), 256, 0, stream>>>(ATTN, SINV, S, T);

    ln(T, I(lb+20), I(lb+21), XLN, ROWS_T);               // ln2
    gemm(XLN, DD, I(lb+22), HD, 0, I(lb+23), nullptr, 1, HIDB, HD, ROWS_T, HD, DD, ACT_GELU);
    gemm(HIDB, HD, I(lb+24), DD, 0, I(lb+25), T, 1, T, DD, ROWS_T, DD, HD, ACT_NONE);
  }

  // ---- mm / ms / sf ----
  trans_sd_k<<<ew((long)NBS*DD*NSQ), 256, 0, stream>>>(T, TT);
  gemm(TT, NSQ, I(109), NP, 0, I(110), nullptr, 1, TMM, NP, NBS*DD, NP, NSQ, ACT_NONE);
  ms_k<<<dim3(((int)((long)NBS*DD) + 7)/8), 256, 0, stream>>>(TMM, I(111), I(112), ORIX);
  trans_dp_k<<<ew((long)NBS*NP*DD), 256, 0, stream>>>(TMM, XP2);
  gemm(XP2, DD, I(113), DD, 0, I(114), ORIX, NP, X1, DD, ROWS_P, DD, DD, ACT_NONE);

  // ---- sisa (n==1 => attention == 1 exactly; q/k/softmax dead) ----
  const float *smu=I(117), *sWv=I(124), *sWih=I(125), *sWhh=I(126);
  const float *sbih=I(127), *sbhh=I(128);
  ln(X1, I(115), I(116), XS1, ROWS_P);                    // sisa outer ln
  ln(XS1, I(118), I(119), XS2, ROWS_P);                   // ln_in
  gemm(XS2, DD, sWv, DD, 0, nullptr, nullptr, 1, VS, DD, ROWS_P, DD, DD, ACT_NONE);
  // gi = v @ Wih^T + bih  -- iteration- and slot-invariant: compute once
  gemm(VS, DD, sWih, DD, 1, sbih, nullptr, 1, GIF, G3, ROWS_P, G3, DD, ACT_NONE);

  for (int c = 0; c < NCHUNK; ++c) {
    long s0 = (long)c * RC;
    const long R8 = (long)RC*NSLOT;
    bcast_row_k<<<ew(R8*DD), 256, 0, stream>>>(smu, SL, R8*DD);
    for (int it = 0; it < 3; ++it) {
      gemm(SL, DD, sWhh, DD, 1, sbhh, nullptr, 1, GHc, G3, (int)R8, G3, DD, ACT_NONE);
      gru_k<<<ew(R8*DD), 256, 0, stream>>>(GIF + s0*G3, NSLOT, GHc, SL, R8);
      ln(SL, I(129), I(130), SLNc, R8);                   // ln_ff
      gemm(SLNc, DD, I(131), HD, 0, I(132), nullptr, 1, HIDc, HD, (int)R8, HD, DD, ACT_RELU);
      gemm(HIDc, HD, I(133), DD, 0, I(134), SL, 1, SL, DD, (int)R8, DD, HD, ACT_NONE);
    }
    slotsum_k<<<ew((long)RC*DD), 256, 0, stream>>>(SL, X1 + s0*DD, Yc, RC);
    ln(Yc, I(135), I(136), Yc, RC);                       // pre-LN (in place, safe)
    gemm(Yc, DD, I(137), HD, 0, I(138), nullptr, 1, HH, HD, RC, HD, DD, ACT_TANH);
    gemm(HH, HD, I(139), NOUT, 0, I(140), nullptr, 1, OB + s0*NOUT, NOUT, RC, NOUT, HD, ACT_NONE);
  }

  outperm_k<<<ew((long)NP*NBS*NOUT), 256, 0, stream>>>(OB, (float*)d_out);
}
